// HyperWeight_81312320848269
// MI455X (gfx1250) — matrix operand order repaired and checked by reading, not compiled, not run on hardware
//
#include <hip/hip_runtime.h>
#include <hip/hip_bf16.h>
#include <cstdint>
#include <cstddef>

// ---------------------------------------------------------------------------
// HyperWeight for MI455X (gfx1250, wave32).
//
// Pipeline (all on `stream`):
//   1. zero node accumulators e[N,256] and degree counts
//   2. node_weight MLP  -> nw[N]           (VALU; 1.3 GFLOP, negligible)
//   3. per-edge Dinv[i] = 1/(nw[u]+nw[v]); degree atomics -> Binv[N]
//   4. er0[E,256] = [edge_rep | 0.5*(nf[u]+nf[v])]  stored bf16 (halves traffic)
//   5. repack hc1_W / hc2_W into WMMA B-fragment order (bf16, L2-resident)
//   6. GEMM1 (er0 @ W1, v_wmma_f32_16x16x32_bf16) with FUSED scatter:
//        e[u] += Binv[u]*xh,  e[v] += Binv[v]*xh   (atomics hit 192MB L2)
//      A slab staged to LDS via async global->LDS copies (ASYNCcnt).
//   7. er1 = sigmoid(Dinv*(e[u]+e[v]) + b1)  -> bf16 (in place over er0)
//   8. zero e[N,128]; GEMM2 + fused scatter
//   9. out = sigmoid(Dinv*(e2[u]+e2[v]) + b2)  -> f32 d_out
//
// Fusing the scatter avoids ever materializing xh[E,256] f32 (saves ~650MB of
// HBM traffic); remaining HBM traffic ~1.0GB -> ~45us floor at 23.3 TB/s,
// while the 63 GFLOP of GEMM is ~25us at bf16 WMMA rates: memory bound.
// ---------------------------------------------------------------------------

#define NN   20000          // nodes
#define NE   320000         // edges
#define HID  128
#define C1   256            // 2*HID

typedef __attribute__((ext_vector_type(16))) __bf16 v16bf;
typedef __attribute__((ext_vector_type(8)))  __bf16 v8bf;
typedef __attribute__((ext_vector_type(8)))  float  v8f;

__device__ __forceinline__ float sigmoidf_(float x) { return 1.0f / (1.0f + expf(-x)); }

// ---------------------------------------------------------------- utilities
__global__ void zero_f32_kernel(float* __restrict__ p, int n) {
  int i = blockIdx.x * blockDim.x + threadIdx.x;
  if (i < n) p[i] = 0.0f;
}

// ----------------------------------------------------- node weight MLP (2.)
// one block of 128 threads per node; thread t owns hidden unit t.
__global__ void node_weight_kernel(const float* __restrict__ x,
                                   const float* __restrict__ proto,
                                   const float* __restrict__ W1,   // [256,128]
                                   const float* __restrict__ b1,   // [128]
                                   const float* __restrict__ W2,   // [128,1]
                                   const float* __restrict__ b2,   // [1]
                                   float* __restrict__ nw) {
  int n = blockIdx.x;
  int t = threadIdx.x;  // 0..127
  const float* xr = x + (size_t)n * HID;
  const float* pr = proto + (size_t)n * HID;
  float acc = b1[t];
#pragma unroll 4
  for (int k = 0; k < HID; ++k) acc = fmaf(xr[k], W1[k * HID + t], acc);
#pragma unroll 4
  for (int k = 0; k < HID; ++k) acc = fmaf(pr[k], W1[(HID + k) * HID + t], acc);
  float h = fmaxf(acc, 0.0f) * W2[t];
  __shared__ float red[HID];
  red[t] = h;
  __syncthreads();
  for (int s = HID / 2; s > 0; s >>= 1) {
    if (t < s) red[t] += red[t + s];
    __syncthreads();
  }
  if (t == 0) nw[n] = sigmoidf_(red[0] + b2[0]);
}

// -------------------------------------------------- per-edge Dinv + degrees
__global__ void edge_aux_kernel(const int* __restrict__ ei, const float* __restrict__ nw,
                                float* __restrict__ Dinv, float* __restrict__ deg) {
  int i = blockIdx.x * blockDim.x + threadIdx.x;
  if (i >= NE) return;
  int u = ei[i], v = ei[NE + i];
  float s = nw[u] + nw[v];
  Dinv[i] = (s > 0.0f) ? (1.0f / s) : 0.0f;
  atomicAdd(deg + u, 1.0f);
  atomicAdd(deg + v, 1.0f);
}

__global__ void binv_kernel(const float* __restrict__ deg, float* __restrict__ Binv) {
  int i = blockIdx.x * blockDim.x + threadIdx.x;
  if (i >= NN) return;
  float d = deg[i];
  Binv[i] = (d > 0.0f) ? (1.0f / d) : 0.0f;
}

// ---------------------------------------------------------- er0 build (4.)
__global__ void build_er0_kernel(const int* __restrict__ ei,
                                 const float* __restrict__ erep,  // [E,192]
                                 const float* __restrict__ nf,    // [N,64]
                                 __bf16* __restrict__ er) {
  int i = blockIdx.x;
  int c = threadIdx.x;  // 0..255
  float val;
  if (c < 192) {
    val = erep[(size_t)i * 192 + c];
  } else {
    int u = ei[i], v = ei[NE + i];
    int cc = c - 192;
    val = 0.5f * (nf[(size_t)u * 64 + cc] + nf[(size_t)v * 64 + cc]);
  }
  er[(size_t)i * C1 + c] = (__bf16)val;
}

// --------------------------------------- weight repack to B-fragment order
// B-fragment (32x16 bf16, wave32): lane holds column N = lane&15;
// its 16 values are K = s*32 + (lane<16 ? 0 : 8) + q + (q<8 ? 0 : 8), q = 0..15
// (B/C/D "row striped across lanes" layout, ISA 7.12.2).
__global__ void repack_w_kernel(const float* __restrict__ W,  // [256, Ncols]
                                __bf16* __restrict__ Wp, int Ncols) {
  int idx = blockIdx.x * blockDim.x + threadIdx.x;
  int NT = Ncols >> 4;
  int total = 8 * NT * 32;  // (k-steps) x (n-tiles) x lanes
  if (idx >= total) return;
  int lane = idx & 31;
  int tile = (idx >> 5) % NT;
  int s    = (idx >> 5) / NT;
  int n  = (tile << 4) + (lane & 15);
  int kb = s * 32 + ((lane < 16) ? 0 : 8);
  __bf16* dst = Wp + ((size_t)(s * NT + tile) * 32 + lane) * 16;
#pragma unroll
  for (int q = 0; q < 16; ++q) dst[q] = (__bf16)W[(size_t)(kb + q + (q < 8 ? 0 : 8)) * Ncols + n];
}

// ------------------------------------- GEMM + fused scatter (6. and 8.)
// Block = one 16-row slab of A (async-staged to LDS), one wave per 16-col tile.
// A-fragment (16x32 bf16): lane holds row M=lane&15; K chunks
// {kb..kb+7, kb+16..kb+23} for lanes 0-15, shifted by 8 for lanes 16-31.
template <int NT>  // number of 16-column tiles (16 -> NC=256, 8 -> NC=128)
__global__ void gemm_scatter_kernel(const __bf16* __restrict__ A,   // [E, 256] bf16
                                    const __bf16* __restrict__ Wp,  // packed B frags
                                    const int* __restrict__ ei,     // [2, E]
                                    const float* __restrict__ Binv, // [N]
                                    float* __restrict__ e) {        // [N, NC]
  constexpr int NC = NT * 16;
  const int m0   = blockIdx.x * 16;
  const int lane = threadIdx.x & 31;
  const int wave = threadIdx.x >> 5;  // = n-tile index

  // +8 bf16 (16B) row pad: fragment reads land on distinct LDS banks
  __shared__ __bf16 As[16][C1 + 8];

  // async stage of the 16x256 bf16 A slab (8KB): one b128 global->LDS copy
  // per thread per iteration, VGPR-file bypassed, tracked by ASYNCcnt.
  for (int idx = threadIdx.x * 8; idx < 16 * C1; idx += NT * 32 * 8) {
    int r = idx >> 8;
    int c = idx & (C1 - 1);
    unsigned lds_addr = (unsigned)(size_t)&As[r][c];
    const __bf16* gsrc = A + (size_t)(m0 + r) * C1 + c;
    asm volatile("global_load_async_to_lds_b128 %0, %1, off"
                 :: "v"(lds_addr), "v"(gsrc) : "memory");
  }
  asm volatile("s_wait_asynccnt 0x0" ::: "memory");
  __syncthreads();

  const int row  = lane & 15;
  const int offA = (lane < 16) ? 0 : 8;
  const __bf16* arow = &As[row][0];

  v8f acc = {0.f, 0.f, 0.f, 0.f, 0.f, 0.f, 0.f, 0.f};
#pragma unroll
  for (int s = 0; s < 8; ++s) {  // K = 256 = 8 * 32
    v8bf a0 = *(const v8bf*)(arow + s * 32 + offA);
    v8bf a1 = *(const v8bf*)(arow + s * 32 + offA + 16);
    v16bf a = __builtin_shufflevector(a0, a1, 0, 1, 2, 3, 4, 5, 6, 7,
                                      8, 9, 10, 11, 12, 13, 14, 15);
    v16bf b = *(const v16bf*)(Wp + ((size_t)(s * NT + wave) * 32 + lane) * 16);
    acc = __builtin_amdgcn_wmma_f32_16x16x32_bf16(false, a, false, b,
                                                  (short)0, acc, false, false);
  }

  // fused hyperconv scatter: e[node] += Binv[node] * xh[edge]
  // C layout: VGPR j -> M = j (lanes 0-15) / j+8 (lanes 16-31), N = lane&15
  const int n     = (wave << 4) + (lane & 15);
  const int rbase = m0 + ((lane < 16) ? 0 : 8);
#pragma unroll
  for (int j = 0; j < 8; ++j) {
    int m = rbase + j;
    int u = ei[m], v = ei[NE + m];
    float val = acc[j];
    atomicAdd(e + (size_t)u * NC + n, Binv[u] * val);
    atomicAdd(e + (size_t)v * NC + n, Binv[v] * val);
  }
}

// ------------------------------------------- gather + sigmoid (7. and 9.)
__global__ void gather_sig_bf16_kernel(const float* __restrict__ e,
                                       const int* __restrict__ ei,
                                       const float* __restrict__ Dinv,
                                       const float* __restrict__ bias,
                                       __bf16* __restrict__ er) {
  int i = blockIdx.x, c = threadIdx.x;  // 256 threads
  int u = ei[i], v = ei[NE + i];
  float val = Dinv[i] * (e[(size_t)u * C1 + c] + e[(size_t)v * C1 + c]) + bias[c];
  er[(size_t)i * C1 + c] = (__bf16)sigmoidf_(val);
}

__global__ void gather_sig_f32_kernel(const float* __restrict__ e,
                                      const int* __restrict__ ei,
                                      const float* __restrict__ Dinv,
                                      const float* __restrict__ bias,
                                      float* __restrict__ out) {
  int i = blockIdx.x, c = threadIdx.x;  // 128 threads
  int u = ei[i], v = ei[NE + i];
  float val = Dinv[i] * (e[(size_t)u * HID + c] + e[(size_t)v * HID + c]) + bias[c];
  out[(size_t)i * HID + c] = sigmoidf_(val);
}

// ---------------------------------------------------------------- launcher
extern "C" void kernel_launch(void* const* d_in, const int* in_sizes, int n_in,
                              void* d_out, int out_size, void* d_ws, size_t ws_size,
                              hipStream_t stream) {
  const int*   ei    = (const int*)d_in[0];    // edge_index [2,E]
  const float* erep  = (const float*)d_in[1];  // edge_rep [E,192]
  const float* x     = (const float*)d_in[2];  // [N,128]
  /* d_in[3] hyper_edge is derived from edge_index; not needed */
  const float* proto = (const float*)d_in[4];  // [N,128]
  const float* nf    = (const float*)d_in[5];  // [N,64]
  const float* l1W   = (const float*)d_in[6];
  const float* l1b   = (const float*)d_in[7];
  const float* l2W   = (const float*)d_in[8];
  const float* l2b   = (const float*)d_in[9];
  const float* W1    = (const float*)d_in[10];  // hc1_W [256,256]
  const float* b1    = (const float*)d_in[11];
  const float* W2    = (const float*)d_in[12];  // hc2_W [256,128]
  const float* b2    = (const float*)d_in[13];
  float* out = (float*)d_out;
  (void)in_sizes; (void)n_in; (void)out_size; (void)ws_size;

  auto align256 = [](size_t o) { return (o + 255) & ~(size_t)255; };
  char* ws = (char*)d_ws;
  size_t off = 0;
  float* nw   = (float*)(ws + off); off = align256(off + (size_t)NN * 4);
  float* Dinv = (float*)(ws + off); off = align256(off + (size_t)NE * 4);
  float* deg  = (float*)(ws + off); off = align256(off + (size_t)NN * 4);
  float* Binv = (float*)(ws + off); off = align256(off + (size_t)NN * 4);
  float* e    = (float*)(ws + off); off = align256(off + (size_t)NN * C1 * 4);
  __bf16* er  = (__bf16*)(ws + off); off = align256(off + (size_t)NE * C1 * 2);
  __bf16* W1p = (__bf16*)(ws + off); off = align256(off + (size_t)8 * 16 * 32 * 16 * 2);
  __bf16* W2p = (__bf16*)(ws + off); off = align256(off + (size_t)8 * 8 * 32 * 16 * 2);

  zero_f32_kernel<<<(NN * C1 + 255) / 256, 256, 0, stream>>>(e, NN * C1);
  zero_f32_kernel<<<(NN + 255) / 256, 256, 0, stream>>>(deg, NN);

  node_weight_kernel<<<NN, HID, 0, stream>>>(x, proto, l1W, l1b, l2W, l2b, nw);
  edge_aux_kernel<<<(NE + 255) / 256, 256, 0, stream>>>(ei, nw, Dinv, deg);
  binv_kernel<<<(NN + 255) / 256, 256, 0, stream>>>(deg, Binv);

  build_er0_kernel<<<NE, C1, 0, stream>>>(ei, erep, nf, er);
  repack_w_kernel<<<(8 * 16 * 32 + 255) / 256, 256, 0, stream>>>(W1, W1p, C1);
  repack_w_kernel<<<(8 * 8 * 32 + 255) / 256, 256, 0, stream>>>(W2, W2p, HID);

  // hyperconv 1: GEMM (N=256) + fused scatter, then gather+sigmoid -> er (bf16)
  gemm_scatter_kernel<16><<<NE / 16, 16 * 32, 0, stream>>>(er, W1p, ei, Binv, e);
  gather_sig_bf16_kernel<<<NE, C1, 0, stream>>>(e, ei, Dinv, b1, er);

  // hyperconv 2: GEMM (N=128) + fused scatter, then gather+sigmoid -> d_out
  zero_f32_kernel<<<(NN * HID + 255) / 256, 256, 0, stream>>>(e, NN * HID);
  gemm_scatter_kernel<8><<<NE / 16, 8 * 32, 0, stream>>>(er, W2p, ei, Binv, e);
  gather_sig_f32_kernel<<<NE, HID, 0, stream>>>(e, ei, Dinv, b2, out);
}
